// GroupedQueryAttention_6571299963258
// MI455X (gfx1250) — matrix operand order repaired and checked by reading, not compiled, not run on hardware
//
#include <hip/hip_runtime.h>

typedef _Float16 h16;
typedef __attribute__((ext_vector_type(4)))  _Float16 v4h;
typedef __attribute__((ext_vector_type(8)))  _Float16 v8h;
typedef __attribute__((ext_vector_type(16))) _Float16 v16h;
typedef __attribute__((ext_vector_type(4)))  float    v4f;
typedef __attribute__((ext_vector_type(8)))  float    v8f;

#define B_   2
#define T_   2048
#define D_   2048
#define H_   32
#define KVH_ 8
#define HD_  64
#define LDP  40   // padded LDS panel pitch in halves (80 B rows, 16B-aligned)

// ---------------------------------------------------------------------------
// Async global->LDS copy of 16 bytes (per lane). VDST = LDS byte offset
// (low 32 bits of the generic pointer == wave-relative LDS address, ISA §10.2),
// VADDR = 64-bit global address. Tracked by ASYNCcnt.
static __device__ __forceinline__ void async_copy_b128(const h16* g, h16* l) {
  const unsigned loff = (unsigned)(unsigned long long)l;
  asm volatile("global_load_async_to_lds_b128 %0, %1, off"
               :: "v"(loff), "v"(g) : "memory");
}

// ---------------------------------------------------------------------------
// WMMA fragment loaders (16x16x32 f16, wave32)
//
// A (16x32, MxK), 16-bit layout per ISA 7.12.2:
//   lane<16 : M=lane,    K = 0..7  in v0..3, K = 16..23 in v4..7
//   lane>=16: M=lane-16, K = 8..15 in v0..3, K = 24..31 in v4..7
static __device__ __forceinline__ v16h load_frag_A(const h16* __restrict__ base,
                                                   int ld, int row0, int col0, int lane) {
  const int m  = lane & 15;
  const int kb = lane >> 4;
  const h16* p = base + (size_t)(row0 + m) * ld + col0 + kb * 8;
  v8h lo = *(const v8h*)p;          // K = kb*8 + 0..7
  v8h hi = *(const v8h*)(p + 16);   // K = kb*8 + 16..23
  v16h r;
#pragma unroll
  for (int u = 0; u < 8; ++u) { r[u] = lo[u]; r[u + 8] = hi[u]; }
  return r;
}

// B (32x16, KxN) mirrored from the ISA sparse-B description:
//   lane<16 : N=lane,    K = 0..7  in v0..3, K = 16..23 in v4..7
//   lane>=16: N=lane-16, K = 8..15 in v0..3, K = 24..31 in v4..7
static __device__ __forceinline__ v16h load_frag_B(const h16* __restrict__ base,
                                                   int ld, int n0, int k0, int lane) {
  const int n  = lane & 15;
  const int kb = lane >> 4;
  const h16* p = base + (size_t)(n0 + n) * ld + k0 + kb * 8;
  v8h lo = *(const v8h*)p;
  v8h hi = *(const v8h*)(p + 16);
  v16h r;
#pragma unroll
  for (int u = 0; u < 8; ++u) { r[u] = lo[u]; r[u + 8] = hi[u]; }
  return r;
}

static __device__ __forceinline__ v8f vzero8f() {
  v8f z;
#pragma unroll
  for (int u = 0; u < 8; ++u) z[u] = 0.0f;
  return z;
}

// ---------------------------------------------------------------------------
// fp32 -> f16 conversion (vectorized x4)
__global__ void f32_to_f16_kernel(const float* __restrict__ src, h16* __restrict__ dst, int n4) {
  int i = blockIdx.x * blockDim.x + threadIdx.x;
  if (i >= n4) return;
  v4f x = *(const v4f*)(src + 4 * (size_t)i);
  v4h y;
#pragma unroll
  for (int u = 0; u < 4; ++u) y[u] = (h16)x[u];
  *(v4h*)(dst + 4 * (size_t)i) = y;
}

// ---------------------------------------------------------------------------
// GEMM: C[M,N] = A[M,K] (row-major f16) x W[N,K]^T (row-major f16)
// 256 thr = 8 waves (2x4); block tile 128x256; wave tile 64x64 (4x4 C tiles).
// K-panels (128x32 A, 256x32 B) double-buffered in LDS, filled with
// global_load_async_to_lds_b128 and drained with s_wait_asynccnt.
template <bool OUT_F32>
__global__ __launch_bounds__(256) void gemm_xwT(const h16* __restrict__ A,
                                                const h16* __restrict__ W,
                                                void* __restrict__ Cout,
                                                int M, int N, int K) {
  __shared__ h16 sA[2][128 * LDP];   // 2 x 10 KB
  __shared__ h16 sB[2][256 * LDP];   // 2 x 20 KB   (60 KB total)

  const int tid  = threadIdx.x;
  const int lane = tid & 31;
  const int wave = tid >> 5;
  const int wm   = wave >> 2;                 // 0..1
  const int wn   = wave & 3;                  // 0..3
  const int m_blk = blockIdx.y * 128;
  const int n_blk = blockIdx.x * 256;

  // staging assignment: thread -> (row = tid>>1, 16-half segment = (tid&1)*16)
  const int sr = tid >> 1;                    // 0..127
  const int sc = (tid & 1) * 16;

  v8f acc[4][4];
#pragma unroll
  for (int i = 0; i < 4; ++i)
#pragma unroll
    for (int j = 0; j < 4; ++j) acc[i][j] = vzero8f();

  auto issue_stage = [&](int buf, int k0) {
    const h16* gA = A + (size_t)(m_blk + sr) * K + (k0 + sc);
    h16*       lA = &sA[buf][sr * LDP + sc];
    async_copy_b128(gA,     lA);
    async_copy_b128(gA + 8, lA + 8);
    const h16* gB0 = W + (size_t)(n_blk + sr) * K + (k0 + sc);
    h16*       lB0 = &sB[buf][sr * LDP + sc];
    async_copy_b128(gB0,     lB0);
    async_copy_b128(gB0 + 8, lB0 + 8);
    const h16* gB1 = gB0 + (size_t)128 * K;
    h16*       lB1 = lB0 + 128 * LDP;
    async_copy_b128(gB1,     lB1);
    async_copy_b128(gB1 + 8, lB1 + 8);
  };

  issue_stage(0, 0);
  const int nk = K >> 5;
  for (int s = 0; s < nk; ++s) {
    const int buf = s & 1;
    if (s + 1 < nk) {
      issue_stage(buf ^ 1, (s + 1) << 5);     // prefetch next panel (async)
      asm volatile("s_wait_asynccnt 6" ::: "memory");  // this stage's 6 done
    } else {
      asm volatile("s_wait_asynccnt 0" ::: "memory");
    }
    __syncthreads();                          // panel `buf` visible to all waves

    v16h a[4];
#pragma unroll
    for (int mi = 0; mi < 4; ++mi)
      a[mi] = load_frag_A(sA[buf], LDP, wm * 64 + mi * 16, 0, lane);
#pragma unroll
    for (int nj = 0; nj < 4; ++nj) {
      v16h b = load_frag_B(sB[buf], LDP, wn * 64 + nj * 16, 0, lane);
#pragma unroll
      for (int mi = 0; mi < 4; ++mi)
        acc[mi][nj] = __builtin_amdgcn_wmma_f32_16x16x32_f16(
            false, a[mi], false, b, (short)0, acc[mi][nj], false, false);
    }
    __syncthreads();                          // done reading before refill
  }

  // C layout: element r of v8f -> row r + 8*(lane>>4), col lane&15
  const int rb = (lane >> 4) * 8;
  const int cn = lane & 15;
#pragma unroll
  for (int mi = 0; mi < 4; ++mi)
#pragma unroll
    for (int nj = 0; nj < 4; ++nj)
#pragma unroll
      for (int r = 0; r < 8; ++r) {
        const size_t row = (size_t)(m_blk + wm * 64 + mi * 16 + rb + r);
        const int    col = n_blk + wn * 64 + nj * 16 + cn;
        const float  v   = acc[mi][nj][r];
        if (OUT_F32) ((float*)Cout)[row * N + col] = v;
        else         ((h16*)Cout)[row * N + col]   = (h16)v;
      }
}

// ---------------------------------------------------------------------------
// RoPE + head transpose: lin[B,T,Hh*64] -> dst[B,Hh,T,64], rotate-half form
__global__ void rope_kernel(const h16* __restrict__ lin, const float* __restrict__ cosp,
                            const float* __restrict__ sinp, h16* __restrict__ dst, int Hh) {
  const int idx = blockIdx.x * blockDim.x + threadIdx.x;
  const int total = B_ * Hh * T_ * 32;
  if (idx >= total) return;
  const int i = idx & 31;
  const int t = (idx >> 5) & (T_ - 1);
  const int r = idx >> 16;            // 5 bits i + 11 bits t
  const int h = r % Hh;
  const int b = r / Hh;
  const size_t src = ((size_t)(b * T_ + t)) * (Hh * HD_) + h * HD_;
  const float x1 = (float)lin[src + i];
  const float x2 = (float)lin[src + 32 + i];
  const float c = cosp[t * 32 + i];
  const float s = sinp[t * 32 + i];
  const size_t d = (((size_t)(b * Hh + h)) * T_ + t) * HD_;
  dst[d + i]      = (h16)(x1 * c - x2 * s);
  dst[d + 32 + i] = (h16)(x2 * c + x1 * s);
}

// V transpose: lin[B,T,KVH*64] -> vT[B,KVH,64,T]
__global__ void vtrans_kernel(const h16* __restrict__ lin, h16* __restrict__ vT) {
  const int idx = blockIdx.x * blockDim.x + threadIdx.x;
  const int total = B_ * KVH_ * HD_ * T_;
  if (idx >= total) return;
  const int t  = idx & (T_ - 1);
  const int d  = (idx >> 11) & 63;
  const int r  = idx >> 17;
  const int kv = r & 7;
  const int b  = r >> 3;
  vT[(((size_t)(b * KVH_ + kv)) * HD_ + d) * T_ + t] =
      lin[((size_t)(b * T_ + t)) * (KVH_ * HD_) + kv * HD_ + d];
}

// ---------------------------------------------------------------------------
// Fused causal flash attention, f16 WMMA, f32 online softmax.
// One wave per 16-row Q tile. grid = (T/16/8, H, B), block = 256 (8 waves).
__global__ __launch_bounds__(256) void attn_kernel(const h16* __restrict__ q,
                                                   const h16* __restrict__ k,
                                                   const h16* __restrict__ vT,
                                                   h16* __restrict__ o) {
  __shared__ float sS[8 * 16 * 32];           // 2 KB per wave, private regions
  const int lane = threadIdx.x & 31;
  const int wave = threadIdx.x >> 5;
  float* sW = sS + wave * 512;

  const int b    = blockIdx.z;
  const int h    = blockIdx.y;
  const int kvh  = h >> 2;                    // N_REP = 4
  const int tile = blockIdx.x * 8 + wave;
  const int t0   = tile * 16;

  const h16* qh = q  + (((size_t)(b * H_   + h))   * T_)  * HD_;
  const h16* kh = k  + (((size_t)(b * KVH_ + kvh)) * T_)  * HD_;
  const h16* vh = vT + (((size_t)(b * KVH_ + kvh)) * HD_) * T_;

  // Q fragments, pre-scaled by 1/sqrt(64) = 0.125 (exact in f16)
  v16h qa[2];
#pragma unroll
  for (int ks = 0; ks < 2; ++ks) {
    qa[ks] = load_frag_A(qh, HD_, t0, 32 * ks, lane);
#pragma unroll
    for (int u = 0; u < 16; ++u) qa[ks][u] = qa[ks][u] * (h16)0.125f;
  }

  const int row   = lane & 15;                // softmax row owned by this lane
  const int rb    = (lane >> 4) * 8;          // C-layout row base
  const int c0    = (lane >> 4) * 8;          // A-layout K base for P packing
  const int t_abs = t0 + row;

  float m_i = -__builtin_inff();
  float l_i = 0.0f;
  v8f o_acc[4];
#pragma unroll
  for (int nj = 0; nj < 4; ++nj) o_acc[nj] = vzero8f();

  for (int s0 = 0; s0 <= t0 + 15; s0 += 32) {
    // S = (Q/8) K^T : 16x32 = two 16x16 C tiles, K-dim 64 = two WMMA steps
    v8f s[2];
    s[0] = vzero8f(); s[1] = vzero8f();
#pragma unroll
    for (int jn = 0; jn < 2; ++jn)
#pragma unroll
      for (int ks = 0; ks < 2; ++ks) {
        v16h bk = load_frag_B(kh, HD_, s0 + 16 * jn, 32 * ks, lane);
        s[jn] = __builtin_amdgcn_wmma_f32_16x16x32_f16(
            false, qa[ks], false, bk, (short)0, s[jn], false, false);
      }

    // prefetch next K/V block while we do softmax (global_prefetch_b8)
    const int s_next = s0 + 32;
    if (s_next <= t0 + 15) {
      __builtin_prefetch(kh + (size_t)(s_next + (lane & 15)) * HD_ + (lane >> 4) * 16, 0, 3);
      __builtin_prefetch(vh + (size_t)(lane & 15) * T_ + s_next + (lane >> 4) * 16, 0, 3);
    }

    // C layout -> LDS (per-wave private; same-wave DS ops are in-order)
#pragma unroll
    for (int jn = 0; jn < 2; ++jn)
#pragma unroll
      for (int r = 0; r < 8; ++r)
        sW[(rb + r) * 32 + jn * 16 + (lane & 15)] = s[jn][r];
    asm volatile("s_wait_dscnt 0" ::: "memory");

    // Each lane reads the 16 values of row (lane&15) it needs for the
    // A-layout P fragment: cols c0+0..7 and c0+16..23.
    float vals[16];
    const float* rowp = sW + row * 32;
#pragma unroll
    for (int q4 = 0; q4 < 2; ++q4) {
      v4f a0 = *(const v4f*)(rowp + c0 + 4 * q4);
      v4f a1 = *(const v4f*)(rowp + c0 + 16 + 4 * q4);
#pragma unroll
      for (int u = 0; u < 4; ++u) { vals[4 * q4 + u] = a0[u]; vals[8 + 4 * q4 + u] = a1[u]; }
    }

    // causal mask + block row-max (combine the two half-wave partials)
    float mb = -__builtin_inff();
#pragma unroll
    for (int u = 0; u < 16; ++u) {
      const int col_abs = s0 + c0 + (u < 8 ? u : (8 + u));
      if (col_abs > t_abs) vals[u] = -__builtin_inff();
      mb = fmaxf(mb, vals[u]);
    }
    mb = fmaxf(mb, __shfl_xor(mb, 16, 32));
    const float m_new = fmaxf(m_i, mb);
    const float alpha = __expf(m_i - m_new);   // first iter: exp(-inf)=0

    float psum = 0.0f;
    v16h pf;
#pragma unroll
    for (int u = 0; u < 16; ++u) {
      const float p = __expf(vals[u] - m_new); // masked -> 0
      psum += p;
      pf[u] = (h16)p;
    }
    psum += __shfl_xor(psum, 16, 32);
    l_i = l_i * alpha + psum;
    m_i = m_new;

    // rescale running O by per-row alpha (row of element r is rb+r)
#pragma unroll
    for (int r = 0; r < 8; ++r) {
      const float ar = __shfl(alpha, rb + r, 32);
#pragma unroll
      for (int nj = 0; nj < 4; ++nj) o_acc[nj][r] *= ar;
    }

    // O += P V : A = P (16x32), B from vT rows (contiguous over s)
#pragma unroll
    for (int nj = 0; nj < 4; ++nj) {
      v16h bv = load_frag_B(vh, T_, nj * 16, s0, lane);
      o_acc[nj] = __builtin_amdgcn_wmma_f32_16x16x32_f16(
          false, pf, false, bv, (short)0, o_acc[nj], false, false);
    }
  }

  // epilogue: divide by row sums, store as [B,T,H,64] f16 (input to W_o GEMM)
#pragma unroll
  for (int r = 0; r < 8; ++r) {
    const float inv = 1.0f / __shfl(l_i, rb + r, 32);
    const int trow  = t0 + rb + r;
#pragma unroll
    for (int nj = 0; nj < 4; ++nj) {
      const float vv = o_acc[nj][r] * inv;
      o[(((size_t)(b * T_ + trow)) * H_ + h) * HD_ + nj * 16 + (lane & 15)] = (h16)vv;
    }
  }
}

// ---------------------------------------------------------------------------
extern "C" void kernel_launch(void* const* d_in, const int* in_sizes, int n_in,
                              void* d_out, int out_size, void* d_ws, size_t ws_size,
                              hipStream_t stream) {
  (void)in_sizes; (void)n_in; (void)out_size; (void)ws_size;
  const float* x    = (const float*)d_in[0];
  const float* cosp = (const float*)d_in[1];
  const float* sinp = (const float*)d_in[2];
  // d_in[3] = mask (causal tril) — computed analytically in attn_kernel
  const float* wq   = (const float*)d_in[4];
  const float* wk   = (const float*)d_in[5];
  const float* wv   = (const float*)d_in[6];
  const float* wo   = (const float*)d_in[7];
  float* out = (float*)d_out;

  const size_t N_X  = (size_t)B_ * T_ * D_;          // 8M
  const size_t N_WQ = (size_t)D_ * D_;               // 4M
  const size_t N_WK = (size_t)KVH_ * HD_ * D_;       // 1M
  const size_t N_QL = (size_t)B_ * T_ * D_;          // 8M
  const size_t N_KL = (size_t)B_ * T_ * KVH_ * HD_;  // 2M

  char* w = (char*)d_ws;
  size_t off = 0;
  h16* xh     = (h16*)(w + off); off += N_X  * 2;
  h16* wqh    = (h16*)(w + off); off += N_WQ * 2;
  h16* wkh    = (h16*)(w + off); off += N_WK * 2;
  h16* wvh    = (h16*)(w + off); off += N_WK * 2;
  h16* woh    = (h16*)(w + off); off += N_WQ * 2;
  h16* q_lin  = (h16*)(w + off); off += N_QL * 2;
  h16* k_lin  = (h16*)(w + off); off += N_KL * 2;
  h16* v_lin  = (h16*)(w + off); off += N_KL * 2;
  h16* q_rope = (h16*)(w + off); off += N_QL * 2;
  h16* k_rope = (h16*)(w + off); off += N_KL * 2;
  h16* vT     = (h16*)(w + off); off += N_KL * 2;
  h16* o_attn = (h16*)(w + off); off += N_QL * 2;   // total 100 MB

  // 1) fp32 -> f16
  auto cvt = [&](const float* s, h16* d, size_t n) {
    const int n4 = (int)(n / 4);
    f32_to_f16_kernel<<<(n4 + 255) / 256, 256, 0, stream>>>(s, d, n4);
  };
  cvt(x, xh, N_X);
  cvt(wq, wqh, N_WQ);
  cvt(wk, wkh, N_WK);
  cvt(wv, wvh, N_WK);
  cvt(wo, woh, N_WQ);

  // 2) QKV projections (WMMA f16 -> f16, LDS-staged, async double-buffered)
  const int M = B_ * T_;
  gemm_xwT<false><<<dim3(D_ / 256, M / 128), 256, 0, stream>>>(xh, wqh, q_lin, M, D_, D_);
  gemm_xwT<false><<<dim3((KVH_ * HD_) / 256, M / 128), 256, 0, stream>>>(xh, wkh, k_lin, M, KVH_ * HD_, D_);
  gemm_xwT<false><<<dim3((KVH_ * HD_) / 256, M / 128), 256, 0, stream>>>(xh, wvh, v_lin, M, KVH_ * HD_, D_);

  // 3) RoPE + transposes
  {
    const int tq = B_ * H_ * T_ * 32;
    rope_kernel<<<(tq + 255) / 256, 256, 0, stream>>>(q_lin, cosp, sinp, q_rope, H_);
    const int tk = B_ * KVH_ * T_ * 32;
    rope_kernel<<<(tk + 255) / 256, 256, 0, stream>>>(k_lin, cosp, sinp, k_rope, KVH_);
    const int tv = B_ * KVH_ * HD_ * T_;
    vtrans_kernel<<<(tv + 255) / 256, 256, 0, stream>>>(v_lin, vT);
  }

  // 4) fused causal flash attention
  attn_kernel<<<dim3(T_ / 16 / 8, H_, B_), 256, 0, stream>>>(q_rope, k_rope, vT, o_attn);

  // 5) output projection (WMMA f16 -> f32 into d_out)
  gemm_xwT<true><<<dim3(D_ / 256, M / 128), 256, 0, stream>>>(o_attn, woh, out, M, D_, D_);
}
